// Module_1_47991964565633
// MI455X (gfx1250) — compile-verified, matrix operand order checked by reading, not run on hardware
//
#include <hip/hip_runtime.h>
#include <math.h>

typedef __attribute__((ext_vector_type(16))) _Float16 v16h;
typedef __attribute__((ext_vector_type(8)))  _Float16 v8h;
typedef __attribute__((ext_vector_type(8)))  float    v8f;
typedef __attribute__((ext_vector_type(4)))  int      v4i;

#define EPSV 1e-5f

#define GLOBAL_AS __attribute__((address_space(1)))
#define SHARED_AS __attribute__((address_space(3)))

// ---- optional CDNA5 async global->LDS copy path (probe-guarded) ----
#if defined(__has_builtin)
#if __has_builtin(__builtin_amdgcn_global_load_async_to_lds_b128)
#define ASYNC_CP 1
#endif
#endif

__device__ __forceinline__ void cp16(const _Float16* g, _Float16* s) {
#ifdef ASYNC_CP
    GLOBAL_AS void* gv = (GLOBAL_AS void*)g;
    SHARED_AS void* sv = (SHARED_AS void*)s;
    __builtin_amdgcn_global_load_async_to_lds_b128(
        (GLOBAL_AS v4i*)gv, (SHARED_AS v4i*)sv, 0, 0);
#else
    *(v8h*)s = *(const v8h*)g;
#endif
}
__device__ __forceinline__ void cp16_wait() {
#ifdef ASYNC_CP
#if __has_builtin(__builtin_amdgcn_s_wait_asynccnt)
    __builtin_amdgcn_s_wait_asynccnt(0);
#else
    asm volatile("s_wait_asynccnt 0x0" ::: "memory");
#endif
#endif
}

// ---- WMMA wave32 fragment helpers (CDNA5 ISA 7.12.2) ----
__device__ __forceinline__ int frag_k(int e, int half) {
    int v = e >> 1, p = e & 1;
    return (v < 4) ? (half * 8 + v * 2 + p)
                   : (16 + half * 8 + (v - 4) * 2 + p);
}

// B fragment = two contiguous 8-half runs: [half*8 .. +7] and [16+half*8 .. +7]
__device__ __forceinline__ v16h load_bfrag(const _Float16* p, int half) {
    v8h lo = *(const v8h*)(p + half * 8);
    v8h hi = *(const v8h*)(p + 16 + half * 8);
    v16h b;
#pragma unroll
    for (int i = 0; i < 8; ++i) { b[i] = lo[i]; b[8 + i] = hi[i]; }
    return b;
}

__device__ __forceinline__ v8f wmma16(v16h a, v16h b, v8f c) {
    return __builtin_amdgcn_wmma_f32_16x16x32_f16(false, a, false, b,
                                                  (short)0, c, false, false);
}

// x_post frame f (0..55) -> base offset (elements) into (nt,cr,h,w) buffers
__device__ __forceinline__ size_t frame_base(int f) {
    int b = f / 7, j = f - b * 7;
    return (size_t)(b * 8 + j + 1) * (size_t)(64 * 784);
}

__device__ __forceinline__ float sampleAt16(const _Float16* __restrict__ p, int iy, int ix) {
    bool valid = (iy >= 0) && (iy < 28) && (ix >= 0) && (ix < 28);
    int cy = min(max(iy, 0), 27), cx = min(max(ix, 0), 27);
    return valid ? (float)p[cy * 28 + cx] : 0.0f;
}

// =====================================================================
// weight f32 -> f16 preconversion
// =====================================================================
__global__ void k_cvt(const float* __restrict__ s, _Float16* __restrict__ d, int n)
{
    int i = blockIdx.x * blockDim.x + threadIdx.x;
    if (i < n) d[i] = (_Float16)s[i];
}

// =====================================================================
// K1: 1x1 conv 256->64 + BN1.  GEMM M=64*784, K=256, N=64.
// 8 waves/block, each wave 16Mx64N (4 wmma per K-step).
// Whole B panel (64x256 f16 = 32KB) staged to LDS once per block.
// =====================================================================
__global__ __launch_bounds__(256) void k_down(
    const float* __restrict__ x, const _Float16* __restrict__ w16,
    const float* __restrict__ bias,
    const float* __restrict__ g, const float* __restrict__ bb,
    const float* __restrict__ mm, const float* __restrict__ vv,
    float* __restrict__ out, _Float16* __restrict__ out16)
{
    __shared__ _Float16 sB[64 * 256];
    const int tid = threadIdx.x;
#pragma unroll
    for (int i = tid; i < 64 * 256 / 8; i += 256)
        cp16(w16 + (size_t)i * 8, sB + (size_t)i * 8);
    cp16_wait();
    __syncthreads();

    const int wave = tid >> 5, lane = tid & 31;
    const int half = lane >> 4, ml = lane & 15;
    int m0 = (blockIdx.x * 8 + wave) * 16;
    int nt = m0 / 784, hw0 = m0 - nt * 784;   // 784 % 16 == 0
    int hwA = hw0 + ml;
    const float* xb = x + (size_t)nt * 256 * 784;
    v8f acc[4] = {};
    for (int kk = 0; kk < 256; kk += 32) {
        v16h a;
#pragma unroll
        for (int e = 0; e < 16; ++e) {
            int k = kk + frag_k(e, half);
            a[e] = (_Float16)xb[(size_t)k * 784 + hwA];
        }
        if (kk + 32 < 256)
            __builtin_prefetch(&xb[(size_t)(kk + 32 + half * 8) * 784 + hwA], 0, 1);
#pragma unroll
        for (int tn = 0; tn < 4; ++tn) {
            v16h b = load_bfrag(sB + (tn * 16 + ml) * 256 + kk, half);
            acc[tn] = wmma16(a, b, acc[tn]);
        }
    }
#pragma unroll
    for (int tn = 0; tn < 4; ++tn) {
        int cr = tn * 16 + ml;
        float s = g[cr] * rsqrtf(vv[cr] + EPSV);
        float t = (bias[cr] - mm[cr]) * s + bb[cr];
        float* ob = out + ((size_t)nt * 64 + cr) * 784;
        _Float16* o16 = out16 + ((size_t)nt * 64 + cr) * 784;
#pragma unroll
        for (int r = 0; r < 8; ++r) {
            float v = acc[tn][r] * s + t;
            int hw = hw0 + r + 8 * half;
            ob[hw] = v;
            o16[hw] = (_Float16)v;
        }
    }
}

// =====================================================================
// K2: 3x3x3 conv3d 64->64 + BNt, reduced over (T,H,W) into cmean[b*8+7].
// GEMM M=8*8*784, K=1728, N=64.  B tile (64x32, 4KB) staged per K-step.
// =====================================================================
__global__ __launch_bounds__(256) void k_tconv(
    const _Float16* __restrict__ out16, const _Float16* __restrict__ w16,
    const float* __restrict__ bias,
    const float* __restrict__ g, const float* __restrict__ bb,
    const float* __restrict__ mm, const float* __restrict__ vv,
    float* __restrict__ cmean)
{
    __shared__ _Float16 sB[64 * 32];
    const int tid = threadIdx.x;
    const int wave = tid >> 5, lane = tid & 31;
    const int half = lane >> 4, ml = lane & 15;
    const int sn = tid >> 2, sq = tid & 3;    // staging row / 16B quarter
    int m0 = (blockIdx.x * 8 + wave) * 16;
    int bI = m0 / 6272; int rem = m0 - bI * 6272;
    int tI = rem / 784; int hw0 = rem - tI * 784;
    int hwA = hw0 + ml; int y = hwA / 28, x0 = hwA - y * 28;
    const _Float16* xb = out16 + (size_t)bI * 8 * 64 * 784;
    v8f acc[4] = {};
    for (int kk = 0; kk < 1728; kk += 32) {
        cp16(w16 + (size_t)sn * 1728 + kk + sq * 8, sB + sn * 32 + sq * 8);
        cp16_wait();
        __syncthreads();
        v16h a;
#pragma unroll
        for (int e = 0; e < 16; ++e) {
            int k = kk + frag_k(e, half);
            int ci = k / 27;  int r27 = k - ci * 27;
            int kd = r27 / 9; int r9  = r27 - kd * 9;
            int kh = r9 / 3;  int kw  = r9 - kh * 3;
            int tt = tI + kd - 1, yy = y + kh - 1, xc = x0 + kw - 1;
            _Float16 av = (_Float16)0.0f;
            if ((unsigned)tt < 8u && (unsigned)yy < 28u && (unsigned)xc < 28u)
                av = xb[((size_t)tt * 64 + ci) * 784 + yy * 28 + xc];
            a[e] = av;
        }
#pragma unroll
        for (int tn = 0; tn < 4; ++tn) {
            v16h b = load_bfrag(sB + (tn * 16 + ml) * 32, half);
            acc[tn] = wmma16(a, b, acc[tn]);
        }
        __syncthreads();
    }
#pragma unroll
    for (int tn = 0; tn < 4; ++tn) {
        int oc = tn * 16 + ml;
        float s = g[oc] * rsqrtf(vv[oc] + EPSV);
        float t = (bias[oc] - mm[oc]) * s + bb[oc];
        float part = 0.0f;
#pragma unroll
        for (int r = 0; r < 8; ++r) part += acc[tn][r] * s + t;
        part += __shfl_xor(part, 16);
        if (lane < 16)
            atomicAdd(&cmean[(bI * 8 + 7) * 64 + oc], part * (1.0f / 6272.0f));
    }
}

// =====================================================================
// Generic direct conv2d (x_post frames) for offset fields (OC=18/50).
// =====================================================================
__global__ void k_conv(const float* __restrict__ out,
                       const float* __restrict__ w, const float* __restrict__ bias,
                       float* __restrict__ dst, int OC, int KS, int pad)
{
    int idx = blockIdx.x * blockDim.x + threadIdx.x;
    int total = 56 * OC * 784;
    if (idx >= total) return;
    int hw = idx % 784; int t2 = idx / 784;
    int oc = t2 % OC;   int f  = t2 / OC;
    const float* xb = out + frame_base(f);
    int y = hw / 28, x = hw - y * 28;
    float acc = bias[oc];
    for (int c = 0; c < 64; ++c) {
        const float* xc = xb + (size_t)c * 784;
        const float* wc = w + ((size_t)oc * 64 + c) * KS * KS;
        for (int ky = 0; ky < KS; ++ky) {
            int yy = y + ky - pad; if ((unsigned)yy >= 28u) continue;
            for (int kx = 0; kx < KS; ++kx) {
                int xx = x + kx - pad; if ((unsigned)xx >= 28u) continue;
                acc += xc[yy * 28 + xx] * wc[ky * KS + kx];
            }
        }
    }
    dst[idx] = acc;
}

// =====================================================================
// 3x3 maxpool on x_post -> mp16 [f, c, hw] (f16, feeds the davg GEMM)
// =====================================================================
__global__ void k_mp(const float* __restrict__ out, _Float16* __restrict__ mp)
{
    int idx = blockIdx.x * blockDim.x + threadIdx.x;
    if (idx >= 56 * 64 * 784) return;
    int hw = idx % 784; int t2 = idx / 784; int c = t2 % 64; int f = t2 / 64;
    const float* xb = out + frame_base(f) + (size_t)c * 784;
    int y = hw / 28, x = hw - y * 28;
    float m = -INFINITY;
    for (int ky = -1; ky <= 1; ++ky) {
        int yy = y + ky; if ((unsigned)yy >= 28u) continue;
        for (int kx = -1; kx <= 1; ++kx) {
            int xx = x + kx; if ((unsigned)xx >= 28u) continue;
            m = fmaxf(m, xb[yy * 28 + xx]);
        }
    }
    mp[idx] = (_Float16)m;
}

// =====================================================================
// Deformable conv einsum: WMMA GEMM, bilinear-gather A loader (f16 src).
// Per M-tile: 16 hw x 64 oc.  B tile staged per K-step.
// =====================================================================
__global__ __launch_bounds__(256) void k_deform(
    const _Float16* __restrict__ out16, const float* __restrict__ off,
    const _Float16* __restrict__ w16, float* __restrict__ dst,
    int KK, int KS, int pad)
{
    __shared__ _Float16 sB[64 * 32];
    const int tid = threadIdx.x;
    const int wave = tid >> 5, lane = tid & 31;
    const int half = lane >> 4, ml = lane & 15;
    const int sn = tid >> 2, sq = tid & 3;
    int tmg = blockIdx.x * 8 + wave;          // 2744 M-tiles
    int f = tmg / 49, tm = tmg - f * 49;
    int hw0 = tm * 16;
    int hwA = hw0 + ml; int y = hwA / 28, x = hwA - y * 28;
    const _Float16* xb = out16 + frame_base(f);
    const float* offf = off + (size_t)f * KK * 2 * 784;
    const int Ktot = 64 * KK;
    v8f acc[4] = {};
    for (int kk = 0; kk < Ktot; kk += 32) {
        cp16(w16 + (size_t)sn * Ktot + kk + sq * 8, sB + sn * 32 + sq * 8);
        cp16_wait();
        __syncthreads();
        v16h a;
#pragma unroll
        for (int e = 0; e < 16; ++e) {
            int k = kk + frag_k(e, half);
            int c = k / KK; int kel = k - c * KK;
            int ky = kel / KS, kx = kel - ky * KS;
            float oy = offf[(kel * 2 + 0) * 784 + hwA];
            float ox = offf[(kel * 2 + 1) * 784 + hwA];
            float py = (float)(y + ky - pad) + oy;
            float px = (float)(x + kx - pad) + ox;
            float fy = floorf(py), fx = floorf(px);
            int y0 = (int)fy, x0i = (int)fx;
            float ly = py - fy, lx = px - fx;
            const _Float16* xc = xb + (size_t)c * 784;
            float v00 = sampleAt16(xc, y0,     x0i);
            float v01 = sampleAt16(xc, y0,     x0i + 1);
            float v10 = sampleAt16(xc, y0 + 1, x0i);
            float v11 = sampleAt16(xc, y0 + 1, x0i + 1);
            float val = v00 * (1.f - ly) * (1.f - lx) + v01 * (1.f - ly) * lx
                      + v10 * ly * (1.f - lx)         + v11 * ly * lx;
            a[e] = (_Float16)val;
        }
#pragma unroll
        for (int tn = 0; tn < 4; ++tn) {
            v16h b = load_bfrag(sB + (tn * 16 + ml) * 32, half);
            acc[tn] = wmma16(a, b, acc[tn]);
        }
        __syncthreads();
    }
#pragma unroll
    for (int tn = 0; tn < 4; ++tn) {
        int oc = tn * 16 + ml;
        float* db = dst + ((size_t)f * 64 + oc) * 784;
#pragma unroll
        for (int r = 0; r < 8; ++r) db[hw0 + r + 8 * half] = acc[tn][r];
    }
}

// =====================================================================
// d_avg: 1x1 conv 64->64 on maxpooled f16 + BN3.  Whole B (8KB) in LDS.
// =====================================================================
__global__ __launch_bounds__(256) void k_davg(
    const _Float16* __restrict__ mp16, const _Float16* __restrict__ w16,
    const float* __restrict__ bias,
    const float* __restrict__ g, const float* __restrict__ bb,
    const float* __restrict__ mm, const float* __restrict__ vv,
    float* __restrict__ dst)
{
    __shared__ _Float16 sB[64 * 64];
    const int tid = threadIdx.x;
#pragma unroll
    for (int i = tid; i < 64 * 64 / 8; i += 256)
        cp16(w16 + (size_t)i * 8, sB + (size_t)i * 8);
    cp16_wait();
    __syncthreads();

    const int wave = tid >> 5, lane = tid & 31;
    const int half = lane >> 4, ml = lane & 15;
    int tmg = blockIdx.x * 8 + wave;
    int f = tmg / 49, tm = tmg - f * 49;
    int hw0 = tm * 16; int hwA = hw0 + ml;
    const _Float16* mb = mp16 + (size_t)f * 64 * 784;
    v8f acc[4] = {};
#pragma unroll
    for (int kk = 0; kk < 64; kk += 32) {
        v16h a;
#pragma unroll
        for (int e = 0; e < 16; ++e) {
            int k = kk + frag_k(e, half);
            a[e] = mb[(size_t)k * 784 + hwA];
        }
#pragma unroll
        for (int tn = 0; tn < 4; ++tn) {
            v16h b = load_bfrag(sB + (tn * 16 + ml) * 64 + kk, half);
            acc[tn] = wmma16(a, b, acc[tn]);
        }
    }
#pragma unroll
    for (int tn = 0; tn < 4; ++tn) {
        int oc = tn * 16 + ml;
        float s = g[oc] * rsqrtf(vv[oc] + EPSV);
        float t = (bias[oc] - mm[oc]) * s + bb[oc];
        float* db = dst + ((size_t)f * 64 + oc) * 784;
#pragma unroll
        for (int r = 0; r < 8; ++r) db[hw0 + r + 8 * half] = acc[tn][r] * s + t;
    }
}

// =====================================================================
// diff spatial mean: cmean[b*8+j, c] = mean_hw |d3+d5+da - 3*x_pre|
// =====================================================================
__global__ void k_diff(const float* __restrict__ d3, const float* __restrict__ d5,
                       const float* __restrict__ da, const float* __restrict__ out,
                       float* __restrict__ cmean)
{
    int f = blockIdx.x / 64; int c = blockIdx.x % 64;
    int b = f / 7, j = f - b * 7;
    const float* xp = out + ((size_t)(b * 8 + j) * 64 + c) * 784;  // x_pre
    size_t base = ((size_t)f * 64 + c) * 784;
    float s = 0.0f;
    for (int i = threadIdx.x; i < 784; i += 256)
        s += fabsf(d3[base + i] + d5[base + i] + da[base + i] - 3.0f * xp[i]);
    __shared__ float red[8];
#pragma unroll
    for (int o = 16; o > 0; o >>= 1) s += __shfl_down(s, o);
    if ((threadIdx.x & 31) == 0) red[threadIdx.x >> 5] = s;
    __syncthreads();
    if (threadIdx.x < 32) {
        float v = (threadIdx.x < 8) ? red[threadIdx.x] : 0.0f;
#pragma unroll
        for (int o = 4; o > 0; o >>= 1) v += __shfl_down(v, o);
        if (threadIdx.x == 0)
            cmean[(b * 8 + j) * 64 + c] = v * (1.0f / 784.0f);
    }
}

// =====================================================================
// gate: 64->256 matvec per sample + BN2 + sigmoid - 0.5
// =====================================================================
__global__ void k_gate(const float* __restrict__ cmean, const float* __restrict__ w,
                       const float* __restrict__ bias,
                       const float* __restrict__ g, const float* __restrict__ bb,
                       const float* __restrict__ mm, const float* __restrict__ vv,
                       float* __restrict__ gate)
{
    int nt = blockIdx.x; int c = threadIdx.x;
    const float* cm = cmean + nt * 64;
    float acc = bias[c];
    for (int k = 0; k < 64; ++k) acc += cm[k] * w[c * 64 + k];
    float s = g[c] * rsqrtf(vv[c] + EPSV);
    float v = (acc - mm[c]) * s + bb[c];
    gate[nt * 256 + c] = 1.0f / (1.0f + expf(-v)) - 0.5f;
}

// =====================================================================
// apply: out = x * (1 + gate)
// =====================================================================
__global__ void k_apply(const float* __restrict__ x, const float* __restrict__ gate,
                        float* __restrict__ outp, int total)
{
    int idx = blockIdx.x * blockDim.x + threadIdx.x;
    if (idx >= total) return;
    int nc = idx / 784; int c = nc % 256; int nt = nc / 256;
    outp[idx] = x[idx] * (1.0f + gate[nt * 256 + c]);
}

extern "C" void kernel_launch(void* const* d_in, const int* in_sizes, int n_in,
                              void* d_out, int out_size, void* d_ws, size_t ws_size,
                              hipStream_t stream)
{
    (void)in_sizes; (void)n_in; (void)out_size; (void)ws_size;
    const float* x       = (const float*)d_in[0];
    const float* w_down  = (const float*)d_in[1];
    const float* b_down  = (const float*)d_in[2];
    const float* bn1_g   = (const float*)d_in[3];
    const float* bn1_b   = (const float*)d_in[4];
    const float* bn1_m   = (const float*)d_in[5];
    const float* bn1_v   = (const float*)d_in[6];
    const float* tconv_w = (const float*)d_in[7];
    const float* tconv_b = (const float*)d_in[8];
    const float* bnt_g   = (const float*)d_in[9];
    const float* bnt_b   = (const float*)d_in[10];
    const float* bnt_m   = (const float*)d_in[11];
    const float* bnt_v   = (const float*)d_in[12];
    const float* off3_w  = (const float*)d_in[13];
    const float* off3_b  = (const float*)d_in[14];
    const float* def3_w  = (const float*)d_in[15];
    const float* off5_w  = (const float*)d_in[16];
    const float* off5_b  = (const float*)d_in[17];
    const float* def5_w  = (const float*)d_in[18];
    const float* conv1_w = (const float*)d_in[19];
    const float* conv1_b = (const float*)d_in[20];
    const float* bn3_g   = (const float*)d_in[21];
    const float* bn3_b   = (const float*)d_in[22];
    const float* bn3_m   = (const float*)d_in[23];
    const float* bn3_v   = (const float*)d_in[24];
    const float* conv_w  = (const float*)d_in[25];
    const float* conv_b  = (const float*)d_in[26];
    const float* bn2_g   = (const float*)d_in[27];
    const float* bn2_b   = (const float*)d_in[28];
    const float* bn2_m   = (const float*)d_in[29];
    const float* bn2_v   = (const float*)d_in[30];
    float* outp = (float*)d_out;

    // workspace carve-up (float units, 256B-aligned chunks)
    float* wsf = (float*)d_ws;
    size_t off = 0;
    auto allocf = [&](size_t n) { float* p = wsf + off; off += (n + 63) & ~(size_t)63; return p; };
    auto alloch = [&](size_t n) { return (_Float16*)allocf((n + 1) / 2); };

    float*    out_b    = allocf((size_t)64 * 64 * 784);
    _Float16* out16    = alloch((size_t)64 * 64 * 784);
    float*    off3_o   = allocf((size_t)56 * 18 * 784);
    float*    off5_o   = allocf((size_t)56 * 50 * 784);
    _Float16* mp16     = alloch((size_t)56 * 64 * 784);
    float*    d3_b     = allocf((size_t)56 * 64 * 784);
    float*    d5_b     = allocf((size_t)56 * 64 * 784);
    float*    da_b     = allocf((size_t)56 * 64 * 784);
    float*    cmean_b  = allocf(64 * 64);
    float*    gate_b   = allocf(64 * 256);
    _Float16* wdown16  = alloch(64 * 256);
    _Float16* tconv16  = alloch((size_t)64 * 64 * 27);
    _Float16* def3w16  = alloch((size_t)64 * 64 * 9);
    _Float16* def5w16  = alloch((size_t)64 * 64 * 25);
    _Float16* conv1w16 = alloch(64 * 64);

    (void)hipMemsetAsync(cmean_b, 0, 64 * 64 * sizeof(float), stream);

    // weight f16 preconversion
    k_cvt<<<(16384 + 255) / 256, 256, 0, stream>>>(w_down, wdown16, 16384);
    k_cvt<<<(110592 + 255) / 256, 256, 0, stream>>>(tconv_w, tconv16, 110592);
    k_cvt<<<(36864 + 255) / 256, 256, 0, stream>>>(def3_w, def3w16, 36864);
    k_cvt<<<(102400 + 255) / 256, 256, 0, stream>>>(def5_w, def5w16, 102400);
    k_cvt<<<(4096 + 255) / 256, 256, 0, stream>>>(conv1_w, conv1w16, 4096);

    // 1) conv_down + bn1 : 3136 M-tiles, 8 per block
    k_down<<<392, 256, 0, stream>>>(x, wdown16, b_down, bn1_g, bn1_b, bn1_m, bn1_v,
                                    out_b, out16);

    // 2) conv3d + bnt reduced into cmean slots b*8+7
    k_tconv<<<392, 256, 0, stream>>>(out16, tconv16, tconv_b, bnt_g, bnt_b, bnt_m, bnt_v,
                                     cmean_b);

    // 3) offset fields (direct conv, small OC)
    {
        int tot3 = 56 * 18 * 784;
        k_conv<<<(tot3 + 255) / 256, 256, 0, stream>>>(out_b, off3_w, off3_b, off3_o, 18, 3, 1);
        int tot5 = 56 * 50 * 784;
        k_conv<<<(tot5 + 255) / 256, 256, 0, stream>>>(out_b, off5_w, off5_b, off5_o, 50, 5, 2);
    }

    // 4) maxpool -> f16
    {
        int tot = 56 * 64 * 784;
        k_mp<<<(tot + 255) / 256, 256, 0, stream>>>(out_b, mp16);
    }

    // 5) deformable convs (WMMA, 2744 M-tiles, 8/block)
    k_deform<<<343, 256, 0, stream>>>(out16, off3_o, def3w16, d3_b, 9, 3, 1);
    k_deform<<<343, 256, 0, stream>>>(out16, off5_o, def5w16, d5_b, 25, 5, 2);

    // 6) d_avg = bn3(conv1(mp))  (WMMA)
    k_davg<<<343, 256, 0, stream>>>(mp16, conv1w16, conv1_b, bn3_g, bn3_b, bn3_m, bn3_v, da_b);

    // 7) diff spatial means into cmean slots b*8+j
    k_diff<<<56 * 64, 256, 0, stream>>>(d3_b, d5_b, da_b, out_b, cmean_b);

    // 8) gate matvec + bn2 + sigmoid - 0.5
    k_gate<<<64, 256, 0, stream>>>(cmean_b, conv_w, conv_b, bn2_g, bn2_b, bn2_m, bn2_v, gate_b);

    // 9) out = x * (1 + gate)
    {
        int tot = 64 * 256 * 784;
        k_apply<<<(tot + 255) / 256, 256, 0, stream>>>(x, gate_b, outp, tot);
    }
}
